// SinkAttention_89395449299393
// MI455X (gfx1250) — matrix operand order repaired and checked by reading, not compiled, not run on hardware
//
#include <hip/hip_runtime.h>

// ---------------------------------------------------------------------------
// Causal attention block for MI455X (gfx1250, wave32).
// - All matrix math: v_wmma_f32_16x16x32_bf16 (f32 accumulate).
// - Attention K/V tiles staged by the Tensor Data Mover (TENSOR_LOAD_TO_LDS,
//   TENSORcnt) with an async-copy fallback; output-GEMM A tile staged with
//   GLOBAL_LOAD_ASYNC_TO_LDS_B128 (ASYNCcnt).
// - f32 inputs converted once via v_cvt_pk_bf16_f32.
// ---------------------------------------------------------------------------

typedef __attribute__((ext_vector_type(16))) __bf16 v16bf;
typedef __attribute__((ext_vector_type(8)))  __bf16 v8bf;
typedef __attribute__((ext_vector_type(8)))  float  v8f;
typedef __attribute__((ext_vector_type(4)))  float  v4f;
typedef __attribute__((ext_vector_type(4)))  unsigned int u32x4;
typedef __attribute__((ext_vector_type(8)))  int i32x8;
typedef __attribute__((ext_vector_type(4)))  int i32x4;

union U16 { v16bf v; v8bf h[2]; };

#define B_DIM 4
#define T_DIM 2048
#define D_DIM 2048
#define H_DIM 16
#define HD_DIM 128
#define M_DIM (B_DIM * T_DIM)   // 8192
#define KC 32

static __device__ inline v8f wmma_bf16(v16bf a, v16bf b, v8f c) {
  return __builtin_amdgcn_wmma_f32_16x16x32_bf16(false, a, false, b, (short)0, c,
                                                 false, false);
}

// CDNA5 async copy: 16B per lane, global -> LDS, tracked by ASYNCcnt.
static __device__ inline void async_copy_b128(void* lds_dst, const void* gsrc) {
  asm volatile("global_load_async_to_lds_b128 %0, %1, off"
               :: "v"((unsigned)(size_t)lds_dst), "v"(gsrc)
               : "memory");
}
static __device__ inline void wait_async0() {
  asm volatile("s_wait_asynccnt 0x0" ::: "memory");
}

// ---------------------------------------------------------------------------
// Tensor Data Mover: DMA a 2D bf16 tile (tile_d1 rows x tile_d0 elems, global
// row stride stride_d0 elems) into LDS at lds (rows packed contiguously).
// D# layout per CDNA5 ISA ch.8: group0 {count, lds_addr, global_addr, type=2},
// group1 {data_size=2B, tensor dims, tile dims, dim0 stride}.
// Toolchain here is the 6-arg clang-23 builtin:
//   (u32x4 g0, i32x8 g1, i32x4 g2, i32x4 g3, i32x8 g4, i32 cpol)
// ---------------------------------------------------------------------------
#if defined(__has_builtin) && __has_builtin(__builtin_amdgcn_tensor_load_to_lds)
#define HAVE_TDM 1
static __device__ inline void tdm_load_2d(void* lds, const void* gaddr,
                                          unsigned tensor_d0, unsigned tensor_d1,
                                          unsigned tile_d0, unsigned tile_d1,
                                          unsigned stride_d0) {
  const unsigned long long ga = (unsigned long long)(size_t)gaddr;
  u32x4 g0;
  g0.x = 1u;                                               // count=1 (valid)
  g0.y = (unsigned)(size_t)lds;                            // lds_addr (bytes)
  g0.z = (unsigned)(ga & 0xFFFFFFFFu);                     // global_addr[31:0]
  g0.w = (unsigned)((ga >> 32) & 0x1FFFFFFu) | (2u << 30); // addr[56:32]|type=2
  i32x8 g1;
  g1[0] = (int)(1u << 16);                                 // data_size=1 (2B)
  g1[1] = (int)(tensor_d0 << 16);                          // tensor_dim0 lo16
  g1[2] = (int)((tensor_d0 >> 16) | (tensor_d1 << 16));    // dim0 hi | dim1 lo
  g1[3] = (int)((tensor_d1 >> 16) | (tile_d0 << 16));      // dim1 hi | tile0
  g1[4] = (int)(tile_d1 & 0xFFFFu);                        // tile1 (tile2=0)
  g1[5] = (int)stride_d0;                                  // dim0 stride lo32
  g1[6] = 0;                                               // stride hi, dim1 stride
  g1[7] = 0;
  const i32x4 gz4 = {0, 0, 0, 0};
  const i32x8 gz8 = {0, 0, 0, 0, 0, 0, 0, 0};
  __builtin_amdgcn_tensor_load_to_lds(g0, g1, gz4, gz4, gz8, 0);
}
static __device__ inline void wait_tensor0() {
  __builtin_amdgcn_s_wait_tensorcnt(0);
}
#else
#define HAVE_TDM 0
#endif

// A-fragment (16x32 bf16), base -> [row0][k0], row stride ldk.
// lanes 0-15: rows 0-15, K {0..7,16..23}; lanes 16-31: rows, K {8..15,24..31}
static __device__ inline v16bf load_a_frag(const __bf16* base, int ldk, int lane) {
  const int row  = lane & 15;
  const int koff = (lane >> 4) << 3;   // 0 or 8
  const __bf16* p = base + row * ldk + koff;
  U16 u;
  u.h[0] = *(const v8bf*)(p);
  u.h[1] = *(const v8bf*)(p + 16);
  return u.v;
}

// B-fragment (32x16 bf16) from a [n][k] tile: lane = column,
// lanes 0-15 hold K {0..7,16..23}, lanes 16-31 hold K {8..15,24..31}.
static __device__ inline v16bf load_b_frag(const __bf16* base, int ldk, int lane) {
  const int col  = lane & 15;
  const int koff = (lane >> 4) << 3;   // 0 or 8
  const __bf16* p = base + col * ldk + koff;
  U16 u;
  u.h[0] = *(const v8bf*)(p);
  u.h[1] = *(const v8bf*)(p + 16);
  return u.v;
}

// ---------------------------------------------------------------------------
// Kernel 1: QKV projection.  C(8192x2048) = x(f32) @ W(f32) in bf16 WMMA.
// grid=(M/128, N/128, 3); 8 waves, each owns a 32x64 strip (2A x 4B frags).
//   z=0,1 (Q,K): out[((b*H+h)*T + t)*HD + hd]     (B,H,T,HD)
//   z=2   (V)  : out[((b*H+h)*HD + hd)*T + t]     (B,H,HD,T)  pre-transposed
// ---------------------------------------------------------------------------
__global__ __launch_bounds__(256) void qkv_gemm_kernel(
    const float* __restrict__ x,
    const float* __restrict__ Wq, const float* __restrict__ Wk,
    const float* __restrict__ Wv,
    __bf16* __restrict__ qws, __bf16* __restrict__ kws, __bf16* __restrict__ vws)
{
  __shared__ __align__(64) __bf16 sA[128 * KC];   // [m][k]
  __shared__ __align__(64) __bf16 sB[128 * KC];   // [n][k] (transposed)

  const int tid  = threadIdx.x;
  const int lane = tid & 31;
  const int wave = tid >> 5;
  const int waveM = wave & 3;          // 4 row-groups of 32
  const int waveN = wave >> 2;         // 2 col-groups of 64
  const int m0   = blockIdx.x * 128;
  const int n0   = blockIdx.y * 128;
  const int z    = blockIdx.z;
  const float* W  = (z == 0) ? Wq : (z == 1) ? Wk : Wv;
  __bf16* outp    = (z == 0) ? qws : (z == 1) ? kws : vws;

  const v8f vzero = {0.f,0.f,0.f,0.f,0.f,0.f,0.f,0.f};
  v8f acc[2][4];
  for (int ai = 0; ai < 2; ++ai)
    for (int j = 0; j < 4; ++j) acc[ai][j] = vzero;

  for (int k0 = 0; k0 < D_DIM; k0 += KC) {
    // Stage A: 128x32 f32 -> bf16 (row-major)
    {
      const int r = tid >> 3;           // 0..31
      const int c = (tid & 7) * 4;      // 0..28
      for (int it = 0; it < 4; ++it) {
        const int row = r + it * 32;
        v4f a4 = *(const v4f*)(x + (size_t)(m0 + row) * D_DIM + k0 + c);
        __bf16* d = sA + row * KC + c;
        d[0] = (__bf16)a4.x; d[1] = (__bf16)a4.y;
        d[2] = (__bf16)a4.z; d[3] = (__bf16)a4.w;
      }
      // Stage B: 32x128 f32 -> bf16, transposed to [n][k]
      const int kk = tid >> 5;          // 0..7
      const int nn = (tid & 31) * 4;    // 0..124
      for (int it = 0; it < 4; ++it) {
        const int krow = kk + it * 8;
        v4f b4 = *(const v4f*)(W + (size_t)(k0 + krow) * D_DIM + n0 + nn);
        sB[(nn + 0) * KC + krow] = (__bf16)b4.x;
        sB[(nn + 1) * KC + krow] = (__bf16)b4.y;
        sB[(nn + 2) * KC + krow] = (__bf16)b4.z;
        sB[(nn + 3) * KC + krow] = (__bf16)b4.w;
      }
      if (k0 + KC < D_DIM) {
        __builtin_prefetch(x + (size_t)(m0 + r) * D_DIM + k0 + KC + c, 0, 1);
        __builtin_prefetch(W + (size_t)(k0 + KC + kk) * D_DIM + n0 + nn, 0, 1);
      }
    }
    __syncthreads();
    // batched fragment loads (one DS clause), then 8 back-to-back WMMAs
    v16bf a0 = load_a_frag(sA + (waveM * 32 +  0) * KC, KC, lane);
    v16bf a1 = load_a_frag(sA + (waveM * 32 + 16) * KC, KC, lane);
    v16bf bfr[4];
    #pragma unroll
    for (int j = 0; j < 4; ++j)
      bfr[j] = load_b_frag(sB + (waveN * 64 + j * 16) * KC, KC, lane);
    #pragma unroll
    for (int j = 0; j < 4; ++j) acc[0][j] = wmma_bf16(a0, bfr[j], acc[0][j]);
    #pragma unroll
    for (int j = 0; j < 4; ++j) acc[1][j] = wmma_bf16(a1, bfr[j], acc[1][j]);
    __syncthreads();
  }

  // Epilogue: C layout -> workspace (bf16)
  const int nloc  = lane & 15;
  const int mbase = (lane >> 4) << 3;
  for (int ai = 0; ai < 2; ++ai) {
    for (int j = 0; j < 4; ++j) {
      for (int i = 0; i < 8; ++i) {
        const int gm = m0 + waveM * 32 + ai * 16 + mbase + i;
        const int gn = n0 + waveN * 64 + j * 16 + nloc;
        const int b  = gm >> 11, t = gm & (T_DIM - 1);
        const int h  = gn >> 7, hd = gn & (HD_DIM - 1);
        size_t idx;
        if (z == 2) idx = ((size_t)((b * H_DIM + h) * HD_DIM + hd)) * T_DIM + t;
        else        idx = ((size_t)((b * H_DIM + h) * T_DIM + t)) * HD_DIM + hd;
        outp[idx] = (__bf16)acc[ai][j][i];
      }
    }
  }
}

// ---------------------------------------------------------------------------
// Kernel 2: flash-style causal attention per (b,h).
// grid = (T/128, B*H); block = 256 (8 waves); wave owns 16 query rows.
// K/V tiles staged by TDM (wave 0 issues, TENSORcnt) or async fallback.
// LDS: K 32KB + V 32KB ([hd][key]) + P strips 32KB = 96KB.
// ---------------------------------------------------------------------------
__global__ __launch_bounds__(256) void attn_kernel(
    const __bf16* __restrict__ qws, const __bf16* __restrict__ kws,
    const __bf16* __restrict__ vws, __bf16* __restrict__ aws)
{
  __shared__ __align__(64) __bf16 sK[128 * 128];   // [key][hd]
  __shared__ __align__(64) __bf16 sV[128 * 128];   // [hd][key]
  __shared__ __align__(64) __bf16 sP[128 * 128];   // 8 wave-private 16x128 strips

  const int tid  = threadIdx.x;
  const int lane = tid & 31;
  const int wave = tid >> 5;
  const int qt = blockIdx.x;                // query tile 0..15
  const int bh = blockIdx.y;                // 0..63
  const int b  = bh >> 4, h = bh & 15;

  const size_t bh_off = (size_t)bh * T_DIM * HD_DIM;
  const __bf16* Qb = qws + bh_off;          // [t][hd]
  const __bf16* Kb = kws + bh_off;          // [t][hd]
  const __bf16* Vb = vws + bh_off;          // [hd][t]
  const int q0 = qt * 128;

  // Q fragments for this wave's 16 rows, straight from global.
  v16bf qfr[4];
  {
    const __bf16* base = Qb + (size_t)(q0 + wave * 16) * HD_DIM;
    #pragma unroll
    for (int kk = 0; kk < 4; ++kk)
      qfr[kk] = load_a_frag(base + kk * 32, HD_DIM, lane);
  }

  const v8f vzero = {0.f,0.f,0.f,0.f,0.f,0.f,0.f,0.f};
  v8f accO[8];
  for (int i = 0; i < 8; ++i) accO[i] = vzero;
  float mrow[8], lrow[8];
  for (int i = 0; i < 8; ++i) { mrow[i] = -1e30f; lrow[i] = 0.f; }

  const float inv_scale = 0.08838834764831845f;   // 1/sqrt(128)

  for (int kt = 0; kt <= qt; ++kt) {
#if HAVE_TDM
    // One DMA descriptor per tile, issued by wave 0 only (TENSORcnt).
    if (wave == 0) {
      // K tile: 128 key-rows x 128 hd-elems, row stride HD, -> sK [key][hd]
      tdm_load_2d(sK, Kb + (size_t)kt * 128 * HD_DIM,
                  /*tensor_d0=*/HD_DIM, /*tensor_d1=*/T_DIM,
                  /*tile_d0=*/HD_DIM, /*tile_d1=*/128, /*stride_d0=*/HD_DIM);
      // V tile: 128 hd-rows x 128 key-elems, row stride T, -> sV [hd][key]
      tdm_load_2d(sV, Vb + (size_t)kt * 128,
                  /*tensor_d0=*/T_DIM, /*tensor_d1=*/HD_DIM,
                  /*tile_d0=*/128, /*tile_d1=*/HD_DIM, /*stride_d0=*/T_DIM);
      wait_tensor0();
    }
#else
    #pragma unroll
    for (int s = 0; s < 8; ++s) {
      const int slot = tid + s * 256;      // 0..2047
      const int row  = slot >> 4;          // 0..127
      const int c8   = (slot & 15) * 8;    // 0..120
      async_copy_b128(sK + row * 128 + c8,
                      Kb + (size_t)(kt * 128 + row) * HD_DIM + c8);
      async_copy_b128(sV + row * 128 + c8,
                      Vb + (size_t)row * T_DIM + kt * 128 + c8);
    }
    wait_async0();
#endif
    __syncthreads();

    // S = Q @ K^T  (16 q-rows x 128 keys per wave)
    v8f sfr[8];
    #pragma unroll
    for (int nf = 0; nf < 8; ++nf) {
      v16bf bfr[4];
      #pragma unroll
      for (int kk = 0; kk < 4; ++kk)
        bfr[kk] = load_b_frag(sK + nf * 16 * 128 + kk * 32, 128, lane);
      v8f c = vzero;
      #pragma unroll
      for (int kk = 0; kk < 4; ++kk) c = wmma_bf16(qfr[kk], bfr[kk], c);
      sfr[nf] = c;
    }

    const int nloc  = lane & 15;
    const int mbase = (lane >> 4) << 3;
    const int qbase = q0 + wave * 16 + mbase;

    // scale + causal mask
    for (int nf = 0; nf < 8; ++nf) {
      const int kg = kt * 128 + nf * 16 + nloc;
      for (int i = 0; i < 8; ++i) {
        float sv = sfr[nf][i] * inv_scale;
        if (kg > qbase + i) sv = -1e30f;
        sfr[nf][i] = sv;
      }
    }

    // online softmax: row max over 8 frags + 16 lanes of the half-wave
    float mnew[8], alpha[8];
    for (int i = 0; i < 8; ++i) {
      float v = sfr[0][i];
      for (int nf = 1; nf < 8; ++nf) v = fmaxf(v, sfr[nf][i]);
      for (int off = 1; off < 16; off <<= 1) v = fmaxf(v, __shfl_xor(v, off, 16));
      mnew[i]  = fmaxf(mrow[i], v);
      alpha[i] = __expf(mrow[i] - mnew[i]);
      mrow[i]  = mnew[i];
    }

    // P = exp(S - m), row sums
    for (int i = 0; i < 8; ++i) {
      float rs = 0.f;
      for (int nf = 0; nf < 8; ++nf) {
        float p = __expf(sfr[nf][i] - mnew[i]);
        sfr[nf][i] = p;
        rs += p;
      }
      for (int off = 1; off < 16; off <<= 1) rs += __shfl_xor(rs, off, 16);
      lrow[i] = lrow[i] * alpha[i] + rs;
    }

    // C-layout -> A-layout via wave-private LDS strip (in-order DS per wave)
    __bf16* pw = sP + wave * 16 * 128;
    for (int nf = 0; nf < 8; ++nf)
      for (int i = 0; i < 8; ++i)
        pw[(mbase + i) * 128 + nf * 16 + nloc] = (__bf16)sfr[nf][i];

    // rescale existing O
    for (int hf = 0; hf < 8; ++hf)
      for (int i = 0; i < 8; ++i)
        accO[hf][i] *= alpha[i];

    // O += P @ V  (batch 4 V-fragments per DS clause)
    #pragma unroll
    for (int kk = 0; kk < 4; ++kk) {
      v16bf pfr = load_a_frag(pw + kk * 32, 128, lane);
      #pragma unroll
      for (int half = 0; half < 2; ++half) {
        v16bf vfr[4];
        #pragma unroll
        for (int j = 0; j < 4; ++j)
          vfr[j] = load_b_frag(sV + (half * 4 + j) * 16 * 128 + kk * 32, 128, lane);
        #pragma unroll
        for (int j = 0; j < 4; ++j)
          accO[half * 4 + j] = wmma_bf16(pfr, vfr[j], accO[half * 4 + j]);
      }
    }
    __syncthreads();
  }

  // normalize and write (B,T,H*HD) bf16 for the output projection
  const int nloc  = lane & 15;
  const int mbase = (lane >> 4) << 3;
  for (int hf = 0; hf < 8; ++hf) {
    for (int i = 0; i < 8; ++i) {
      const int t  = q0 + wave * 16 + mbase + i;
      const int hd = hf * 16 + nloc;
      const float o = accO[hf][i] / lrow[i];
      aws[((size_t)(b * T_DIM + t)) * D_DIM + h * HD_DIM + hd] = (__bf16)o;
    }
  }
}

// ---------------------------------------------------------------------------
// Kernel 3: output projection.  out(f32) = attn(bf16) @ Wo(f32->bf16)
// A-tile staged with async global->LDS (already bf16).
// ---------------------------------------------------------------------------
__global__ __launch_bounds__(256) void out_gemm_kernel(
    const __bf16* __restrict__ A, const float* __restrict__ W,
    float* __restrict__ out)
{
  __shared__ __align__(64) __bf16 sA[128 * KC];
  __shared__ __align__(64) __bf16 sB[128 * KC];

  const int tid  = threadIdx.x;
  const int lane = tid & 31;
  const int wave = tid >> 5;
  const int waveM = wave & 3;
  const int waveN = wave >> 2;
  const int m0   = blockIdx.x * 128;
  const int n0   = blockIdx.y * 128;

  const v8f vzero = {0.f,0.f,0.f,0.f,0.f,0.f,0.f,0.f};
  v8f acc[2][4];
  for (int ai = 0; ai < 2; ++ai)
    for (int j = 0; j < 4; ++j) acc[ai][j] = vzero;

  for (int k0 = 0; k0 < D_DIM; k0 += KC) {
    // Stage A (bf16) via async copy: 128x32 = 512 16B chunks, 2/thread
    #pragma unroll
    for (int s = 0; s < 2; ++s) {
      const int slot = tid + s * 256;
      const int row  = slot >> 2;
      const int c8   = (slot & 3) * 8;
      async_copy_b128(sA + row * KC + c8,
                      A + (size_t)(m0 + row) * D_DIM + k0 + c8);
    }
    // Stage B: 32x128 f32 -> bf16 transposed [n][k]
    {
      const int kk = tid >> 5;
      const int nn = (tid & 31) * 4;
      for (int it = 0; it < 4; ++it) {
        const int krow = kk + it * 8;
        v4f b4 = *(const v4f*)(W + (size_t)(k0 + krow) * D_DIM + n0 + nn);
        sB[(nn + 0) * KC + krow] = (__bf16)b4.x;
        sB[(nn + 1) * KC + krow] = (__bf16)b4.y;
        sB[(nn + 2) * KC + krow] = (__bf16)b4.z;
        sB[(nn + 3) * KC + krow] = (__bf16)b4.w;
      }
    }
    wait_async0();
    __syncthreads();
    v16bf a0 = load_a_frag(sA + (waveM * 32 +  0) * KC, KC, lane);
    v16bf a1 = load_a_frag(sA + (waveM * 32 + 16) * KC, KC, lane);
    v16bf bfr[4];
    #pragma unroll
    for (int j = 0; j < 4; ++j)
      bfr[j] = load_b_frag(sB + (waveN * 64 + j * 16) * KC, KC, lane);
    #pragma unroll
    for (int j = 0; j < 4; ++j) acc[0][j] = wmma_bf16(a0, bfr[j], acc[0][j]);
    #pragma unroll
    for (int j = 0; j < 4; ++j) acc[1][j] = wmma_bf16(a1, bfr[j], acc[1][j]);
    __syncthreads();
  }

  const int nloc  = lane & 15;
  const int mbase = (lane >> 4) << 3;
  for (int ai = 0; ai < 2; ++ai)
    for (int j = 0; j < 4; ++j)
      for (int i = 0; i < 8; ++i) {
        const int gm = m0 + waveM * 32 + ai * 16 + mbase + i;
        const int gn = n0 + waveN * 64 + j * 16 + nloc;
        out[(size_t)gm * D_DIM + gn] = acc[ai][j][i];
      }
}

// ---------------------------------------------------------------------------
extern "C" void kernel_launch(void* const* d_in, const int* in_sizes, int n_in,
                              void* d_out, int out_size, void* d_ws, size_t ws_size,
                              hipStream_t stream) {
  const float* x  = (const float*)d_in[0];
  const float* Wq = (const float*)d_in[1];
  const float* Wk = (const float*)d_in[2];
  const float* Wv = (const float*)d_in[3];
  const float* Wo = (const float*)d_in[4];

  // Workspace: Q, K, V, attn-out, each B*H*T*HD bf16 = 32MB (128MB total).
  const size_t PER = (size_t)B_DIM * H_DIM * T_DIM * HD_DIM;  // 16,777,216
  __bf16* qws = (__bf16*)d_ws;
  __bf16* kws = qws + PER;
  __bf16* vws = kws + PER;
  __bf16* aws = vws + PER;

  qkv_gemm_kernel<<<dim3(M_DIM / 128, D_DIM / 128, 3), 256, 0, stream>>>(
      x, Wq, Wk, Wv, qws, kws, vws);
  attn_kernel<<<dim3(T_DIM / 128, B_DIM * H_DIM), 256, 0, stream>>>(
      qws, kws, vws, aws);
  out_gemm_kernel<<<dim3(M_DIM / 128, D_DIM / 128), 256, 0, stream>>>(
      aws, Wo, (float*)d_out);
}
